// AttCML_87969520157214
// MI455X (gfx1250) — compile-verified, matrix operand order checked by reading, not run on hardware
//
#include <hip/hip_runtime.h>
#include <hip/hip_bf16.h>

typedef __attribute__((ext_vector_type(16))) _Float16 v16h;
typedef __attribute__((ext_vector_type(8)))  float    v8f;

#define DIMS   128   // embedding dim
#define PMAX   50    // prefs per sample
#define PPAD   64    // padded to 4 M-tiles of 16
#define LSTR   132   // LDS row stride in floats (132*4=528B: 16B aligned, bank-skewed)

__global__ __launch_bounds__(128)
void AttCML_kernel(const float* __restrict__ user_emb,
                   const float* __restrict__ item_emb,
                   const int*   __restrict__ user_ids,
                   const int*   __restrict__ pos_ids,
                   const int*   __restrict__ neg_ids,
                   const int*   __restrict__ pref_ids,
                   const float* __restrict__ n_prefs,
                   float* __restrict__ out,
                   int B, int n_items)
{
    __shared__ __align__(16) float sPref[PPAD * LSTR];  // 50 real rows, 14 zero rows
    __shared__ __align__(16) float sT[2 * LSTR];        // [0]=pos target, [1]=neg target
    __shared__ float sW[2 * PPAD];                      // raw dot products (pos,neg)
    __shared__ float sAtt[2 * PPAD];                    // attention weights
    __shared__ float sRed[8];                           // cross-wave reduction

    const int b    = blockIdx.x;
    const int tid  = threadIdx.x;
    const int wave = tid >> 5;
    const int lane = tid & 31;

    // ---------------- Phase A: gather pref rows + targets into LDS -------------
    // Each wave loads whole 512B rows: 32 lanes x float4 = 128 floats (coalesced).
    for (int r = wave; r < PPAD; r += 4) {
        float4 v = make_float4(0.f, 0.f, 0.f, 0.f);
        if (r < PMAX) {
            int id = pref_ids[b * PMAX + r];
            if (id < n_items)   // id == n_items is the zero-padding row
                v = *(const float4*)(item_emb + (size_t)id * DIMS + lane * 4);
        }
        *(float4*)(&sPref[r * LSTR + lane * 4]) = v;
    }
    if (wave == 0) {
        int id = pos_ids[b];
        *(float4*)(&sT[0 * LSTR + lane * 4]) =
            *(const float4*)(item_emb + (size_t)id * DIMS + lane * 4);
    } else if (wave == 1) {
        int id = neg_ids[b];
        *(float4*)(&sT[1 * LSTR + lane * 4]) =
            *(const float4*)(item_emb + (size_t)id * DIMS + lane * 4);
    }
    __syncthreads();

    // ---------------- Phase B: W(64x16) = Pref(64x128) x T(128x16) via WMMA ----
    // Wave w owns M-tile rows [16w, 16w+16). Columns: 0=pos target, 1=neg, rest 0.
    {
        const int mrow  = wave * 16 + (lane & 15);
        const int half8 = (lane >> 4) ? 8 : 0;   // lane<16: K 0..7/16..23 ; lane>=16: K 8..15/24..31
        const int col   = lane & 15;
        const float* arow = &sPref[mrow * LSTR];
        const float* brow = &sT[(col < 2 ? col : 0) * LSTR];
        const bool   bval = (col < 2);

        v8f c = {0.f, 0.f, 0.f, 0.f, 0.f, 0.f, 0.f, 0.f};
        #pragma unroll
        for (int kb = 0; kb < DIMS; kb += 32) {
            const int k0 = kb + half8;        // K pairs for VGPRs 0..3
            const int k1 = kb + 16 + half8;   // K pairs for VGPRs 4..7
            v16h a, bm;
            #pragma unroll
            for (int j = 0; j < 4; ++j) {
                float2 fa0 = *(const float2*)(arow + k0 + 2 * j);
                float2 fa1 = *(const float2*)(arow + k1 + 2 * j);
                a[2 * j]     = (_Float16)fa0.x;
                a[2 * j + 1] = (_Float16)fa0.y;
                a[8 + 2 * j]     = (_Float16)fa1.x;
                a[8 + 2 * j + 1] = (_Float16)fa1.y;
                float2 fb0 = *(const float2*)(brow + k0 + 2 * j);
                float2 fb1 = *(const float2*)(brow + k1 + 2 * j);
                bm[2 * j]     = (_Float16)(bval ? fb0.x : 0.f);
                bm[2 * j + 1] = (_Float16)(bval ? fb0.y : 0.f);
                bm[8 + 2 * j]     = (_Float16)(bval ? fb1.x : 0.f);
                bm[8 + 2 * j + 1] = (_Float16)(bval ? fb1.y : 0.f);
            }
            // (neg_a, A, neg_b, B, c_mod, C, reuse_a, reuse_b)
            c = __builtin_amdgcn_wmma_f32_16x16x32_f16(false, a, false, bm,
                                                       (short)0, c, false, false);
        }
        // C layout: VGPR g -> M = 16w + g + (lane<16 ? 0 : 8), N = lane&15
        if (col < 2) {
            #pragma unroll
            for (int g = 0; g < 8; ++g)
                sW[col * PPAD + wave * 16 + half8 + g] = c[g];
        }
    }
    __syncthreads();

    // ---------------- Phase C: masked exp + normalize (wave0=pos, wave1=neg) ---
    if (wave < 2) {
        const float np1 = n_prefs[b] + 1.0f;     // tf.sequence_mask(n_prefs+1, P)
        const int p0 = lane, p1 = lane + 32;
        float e0 = ((float)p0 < np1) ? __expf(sW[wave * PPAD + p0]) : 0.f;
        float e1 = (p1 < PMAX && (float)p1 < np1) ? __expf(sW[wave * PPAD + p1]) : 0.f;
        float s = e0 + e1;
        #pragma unroll
        for (int off = 16; off > 0; off >>= 1) s += __shfl_xor(s, off, 32);
        const float inv = 1.0f / s;              // BETA == 1.0
        sAtt[wave * PPAD + p0] = e0 * inv;
        if (p1 < PMAX) sAtt[wave * PPAD + p1] = e1 * inv;
    }
    __syncthreads();

    // ---------------- Phase D: att-weighted sum, distances, reduce -------------
    {
        const int d = tid;                        // 128 threads <-> 128 dims
        float accP = 0.f, accN = 0.f;
        #pragma unroll 5
        for (int p = 0; p < PMAX; ++p) {
            const float pv = sPref[p * LSTR + d];
            accP = fmaf(sAtt[p], pv, accP);
            accN = fmaf(sAtt[PPAD + p], pv, accN);
        }
        const float u  = user_emb[(size_t)user_ids[b] * DIMS + d];
        const float dp = u + accP - sT[d];
        const float dn = u + accN - sT[LSTR + d];
        float sqP = dp * dp, sqN = dn * dn;
        #pragma unroll
        for (int off = 16; off > 0; off >>= 1) {
            sqP += __shfl_xor(sqP, off, 32);
            sqN += __shfl_xor(sqN, off, 32);
        }
        if (lane == 0) { sRed[wave] = sqP; sRed[4 + wave] = sqN; }
    }
    __syncthreads();
    if (tid == 0) {
        out[b]     = sRed[0] + sRed[1] + sRed[2] + sRed[3];
        out[B + b] = sRed[4] + sRed[5] + sRed[6] + sRed[7];
    }
}

extern "C" void kernel_launch(void* const* d_in, const int* in_sizes, int n_in,
                              void* d_out, int out_size, void* d_ws, size_t ws_size,
                              hipStream_t stream) {
    (void)n_in; (void)out_size; (void)d_ws; (void)ws_size;
    const float* user_emb = (const float*)d_in[0];
    const float* item_emb = (const float*)d_in[1];
    const int*   user_ids = (const int*)d_in[2];
    const int*   pos_ids  = (const int*)d_in[3];
    const int*   neg_ids  = (const int*)d_in[4];
    const int*   pref_ids = (const int*)d_in[5];
    const float* n_prefs  = (const float*)d_in[6];
    float* out = (float*)d_out;

    const int B       = in_sizes[2];
    const int n_items = in_sizes[1] / DIMS;

    AttCML_kernel<<<B, 128, 0, stream>>>(user_emb, item_emb, user_ids, pos_ids,
                                         neg_ids, pref_ids, n_prefs, out, B, n_items);
}
